// BCT_P_74612171866193
// MI455X (gfx1250) — matrix operand order repaired and checked by reading, not compiled, not run on hardware
//
#include <hip/hip_runtime.h>
#include <hip/hip_bf16.h>
#include <stdint.h>

// ---------------------------------------------------------------------------
// Shapes (from reference): C=512,H=4,W=4,SPLIT=4 -> WC=1, L=2048, B=16.
// reverse_inside / flip_inside act on a size-1 axis -> identity.
// 30 sequential 2-layer LSTM steps, each layer = 2 GEMMs [16x2048]@[2048x8192]
// -> weight-streaming bound (~165us floor at 23.3 TB/s with bf16 weights).
// Weights fp32->bf16 once; GEMMs via v_wmma_f32_16x16x32_bf16 (M=16 == tile).
// Step kernel: 8-way K-split per j-tile (1024 waves); fully unrolled K-slice
// with an EXPLICIT depth-1 software pipeline (cur/nxt fragment sets) so
// iteration i+1's 20 b128 loads are issued in source order before iteration
// i's WMMAs -> deep load pipelines; all addresses are base+imm (no per-iter
// address VALU / hazard nops). LDS reduction + fused LSTM cell update.
// ---------------------------------------------------------------------------

typedef __attribute__((ext_vector_type(16))) __bf16 v16bf;
typedef __attribute__((ext_vector_type(8)))  __bf16 v8bf;
typedef __attribute__((ext_vector_type(8)))  float  v8f;

namespace {
constexpr int kL   = 2048;          // L = C*H*WC
constexpr int kG   = 8192;          // 4*L gate width
constexpr int kB   = 16;            // batch
constexpr int kBL  = kB * kL;       // 32768 elems of a [16,2048] buffer
constexpr int kImg = kB * 512 * 16; // 131072 elems of a [16,512,4,4] image
constexpr int kNS  = 8;             // K slices per j-tile (waves per block)
constexpr int kKS  = kL / kNS;      // 256: K range per wave
constexpr int kIt  = kKS / 32;      // 8 WMMA iterations per wave
}

// ---------------------------- device helpers -------------------------------

__device__ __forceinline__ float sigm_(float x) {
  return 1.0f / (1.0f + __expf(-x));
}
__device__ __forceinline__ float tanh_(float x) {
  float t = __expf(-2.0f * fabsf(x));
  float r = (1.0f - t) / (1.0f + t);
  return copysignf(r, x);
}

// A-matrix 16x32 bf16 fragment: this lane's 16 contiguous K values
// (lanes 0-15: K {0..15}; lanes 16-31: K {16..31};
//  caller pre-offsets the pointer by the lane's first chunk).
__device__ __forceinline__ v16bf load_a_frag(const __bf16* p) {
  v8bf lo = *(const v8bf*)(p);
  v8bf hi = *(const v8bf*)(p + 8);
  v16bf r;
#pragma unroll
  for (int i = 0; i < 8; ++i) { r[i] = lo[i]; r[i + 8] = hi[i]; }
  return r;
}

// B-matrix 32x16 bf16 fragment: lane n holds 16 contiguous K values of
// weight row (jbase+n); lanes 16-31 hold K+16 (caller pre-offsets pointer).
__device__ __forceinline__ v16bf load_b_frag(const __bf16* p) {
  v8bf lo = *(const v8bf*)(p);
  v8bf hi = *(const v8bf*)(p + 8);
  v16bf r;
#pragma unroll
  for (int i = 0; i < 8; ++i) { r[i] = lo[i]; r[i + 8] = hi[i]; }
  return r;
}

#define WMMA_BF16(A, B, C)                                                  \
  __builtin_amdgcn_wmma_f32_16x16x32_bf16(false, (A), false, (B), (short)0, \
                                          (C), false, false)

// One iteration's worth of operand fragments (10 x 8 VGPRs).
struct Frags {
  v16bf ax, ah;
  v16bf bi0, bi1, bi2, bi3;
  v16bf bh0, bh1, bh2, bh3;
};

__device__ __forceinline__ Frags
load_frags(const __bf16* xrow, const __bf16* hrow,
           const __bf16* w0, const __bf16* w1,
           const __bf16* w2, const __bf16* w3,
           const __bf16* u0, const __bf16* u1,
           const __bf16* u2, const __bf16* u3, int k) {
  Frags f;
  f.ax  = load_a_frag(xrow + k);
  f.ah  = load_a_frag(hrow + k);
  f.bi0 = load_b_frag(w0 + k);
  f.bi1 = load_b_frag(w1 + k);
  f.bi2 = load_b_frag(w2 + k);
  f.bi3 = load_b_frag(w3 + k);
  f.bh0 = load_b_frag(u0 + k);
  f.bh1 = load_b_frag(u1 + k);
  f.bh2 = load_b_frag(u2 + k);
  f.bh3 = load_b_frag(u3 + k);
  return f;
}

// ------------------------------- kernels -----------------------------------

// fp32 -> bf16 bulk convert (weights), 4 elems/thread.
__global__ void k_cvt_bf16(const float* __restrict__ src,
                           __bf16* __restrict__ dst, int n) {
  int i = (blockIdx.x * blockDim.x + threadIdx.x) * 4;
  if (i + 3 < n) {
    float4 v = *(const float4*)(src + i);
    union { __bf16 h[4]; uint2 u; } pk;
    pk.h[0] = (__bf16)v.x; pk.h[1] = (__bf16)v.y;
    pk.h[2] = (__bf16)v.z; pk.h[3] = (__bf16)v.w;
    *(uint2*)(dst + i) = pk.u;
  } else {
    for (; i < n; ++i) dst[i] = (__bf16)src[i];
  }
}

__global__ void k_zero(uint32_t* __restrict__ p, int n) {
  int i = blockIdx.x * blockDim.x + threadIdx.x;
  if (i < n) p[i] = 0u;
}

// chunk s of x[b,c,h,w] -> bf16 [16,2048]:  out[b*2048 + c*4+h] = x[...,w=s]
__global__ void k_extract(const float* __restrict__ x, int s,
                          __bf16* __restrict__ out) {
  int i = blockIdx.x * blockDim.x + threadIdx.x;
  if (i < kBL) out[i] = (__bf16)x[i * 4 + s];
}

// One LSTM layer-step, fully fused:
//   g = xin@Wih^T + bih + hin@Whh^T + bhh   (g: [16, 8192])
//   i,f,gg,o = split(g); cn = sig(f)*c + sig(i)*tanh(gg); hn = sig(o)*tanh(cn)
// Block = 8 waves = one 16-wide j-tile; wave w owns K in [w*256, w*256+256)
// and all 4 gate tiles (j, j+L, j+2L, j+3L). Fully unrolled, explicitly
// software-pipelined K-slice; partials reduced through LDS; wave 0 applies
// the cell update.
__global__ __launch_bounds__(256, 1)
void k_lstm_step(const __bf16* __restrict__ xin,   // [16,2048] bf16
                 const __bf16* __restrict__ hin,   // [16,2048] bf16
                 const __bf16* __restrict__ Wih,   // [8192,2048] bf16
                 const __bf16* __restrict__ Whh,   // [8192,2048] bf16
                 const float*  __restrict__ bih,   // [8192]
                 const float*  __restrict__ bhh,   // [8192]
                 float*        __restrict__ cbuf,  // [16,2048] in/out (tile-excl.)
                 __bf16*       __restrict__ hout,  // [16,2048] bf16 (dbl-buffered)
                 float*        __restrict__ save)  // optional fp32 copy of hn
{
  __shared__ float red[kNS][4][8][32];   // [kslice][gate][row][lane] = 32 KB

  const int tid   = threadIdx.x;
  const int wid   = tid >> 5;            // 0..7: K slice
  const int lane  = tid & 31;
  const int jbase = blockIdx.x * 16;     // j tile
  const int half  = lane >> 4;
  const int nl    = lane & 15;
  const int k0    = wid * kKS;

  const __bf16* xrow = xin + nl * kL + half * 16 + k0;  // A: lane = batch row
  const __bf16* hrow = hin + nl * kL + half * 16 + k0;

  const __bf16* wi[4];
  const __bf16* wh[4];
#pragma unroll
  for (int g = 0; g < 4; ++g) {                         // B: lane = output col
    size_t row = (size_t)(g * kL + jbase + nl) * kL;
    wi[g] = Wih + row + half * 16 + k0;
    wh[g] = Whh + row + half * 16 + k0;
  }

  v8f acc[4] = {};

  // Explicit depth-1 pipeline inside a fully unrolled loop: the loads for
  // iteration it+1 appear (in source order) before the WMMAs of iteration it,
  // so each wave keeps ~20 b128 loads in flight; all offsets are immediates.
  Frags cur = load_frags(xrow, hrow, wi[0], wi[1], wi[2], wi[3],
                         wh[0], wh[1], wh[2], wh[3], 0);
#pragma unroll
  for (int it = 0; it < kIt; ++it) {
    Frags nxt;
    if (it + 1 < kIt)
      nxt = load_frags(xrow, hrow, wi[0], wi[1], wi[2], wi[3],
                       wh[0], wh[1], wh[2], wh[3], (it + 1) * 32);
    acc[0] = WMMA_BF16(cur.ax, cur.bi0, acc[0]);
    acc[1] = WMMA_BF16(cur.ax, cur.bi1, acc[1]);
    acc[2] = WMMA_BF16(cur.ax, cur.bi2, acc[2]);
    acc[3] = WMMA_BF16(cur.ax, cur.bi3, acc[3]);
    acc[0] = WMMA_BF16(cur.ah, cur.bh0, acc[0]);
    acc[1] = WMMA_BF16(cur.ah, cur.bh1, acc[1]);
    acc[2] = WMMA_BF16(cur.ah, cur.bh2, acc[2]);
    acc[3] = WMMA_BF16(cur.ah, cur.bh3, acc[3]);
    if (it + 1 < kIt) cur = nxt;      // erased by renaming under full unroll
  }

  // Cross-slice reduction through LDS.
#pragma unroll
  for (int g = 0; g < 4; ++g)
#pragma unroll
    for (int r = 0; r < 8; ++r)
      red[wid][g][r][lane] = acc[g][r];
  __syncthreads();

  if (wid == 0) {
    const int j  = jbase + nl;
    const float bI = bih[j]          + bhh[j];
    const float bF = bih[kL + j]     + bhh[kL + j];
    const float bG = bih[2 * kL + j] + bhh[2 * kL + j];
    const float bO = bih[3 * kL + j] + bhh[3 * kL + j];
#pragma unroll
    for (int r = 0; r < 8; ++r) {
      float t[4];
#pragma unroll
      for (int g = 0; g < 4; ++g) {
        float s = 0.0f;
#pragma unroll
        for (int w = 0; w < kNS; ++w) s += red[w][g][r][lane];
        t[g] = s;
      }
      int m   = r + half * 8;              // C/D layout: VGPR r, lane half
      int idx = m * kL + j;
      float gi = t[0] + bI;
      float gf = t[1] + bF;
      float gg = t[2] + bG;
      float go = t[3] + bO;
      float cp = cbuf[idx];
      float cn = sigm_(gf) * cp + sigm_(gi) * tanh_(gg);
      float hn = sigm_(go) * tanh_(cn);
      cbuf[idx] = cn;
      hout[idx] = (__bf16)hn;
      if (save) save[idx] = hn;
    }
  }
}

// x1_parts[1], x1_parts[2], x2_parts[0], x2_parts[1] from the 16 saved chunks.
__device__ __constant__ int c_slotmap[16] = {
    0, 1, 2, 3,      // sec2: de1, p1_0..p1_2
    4, 5, 6, 7,      // sec3: p1_3..p1_6
    15, 14, 13, 12,  // sec4: p2_6..p2_3
    11, 10, 9, 8};   // sec5: p2_2..p2_0, de2

__global__ void k_assemble(const float* __restrict__ saves,
                           float* __restrict__ out) {
  int idx = blockIdx.x * blockDim.x + threadIdx.x;
  if (idx >= 4 * kImg) return;
  int sec = idx >> 17;             // 0..3 -> d_out sections 2..5
  int e   = idx & (kImg - 1);      // [b, c, h, w]
  int w = e & 3;
  int h = (e >> 2) & 3;
  int c = (e >> 4) & 511;
  int b = e >> 13;
  int slot = c_slotmap[sec * 4 + w];
  out[(size_t)(2 + sec) * kImg + e] = saves[slot * kBL + b * kL + c * 4 + h];
}

// out = leaky_relu(conv_W @ concat(x1_parts[1], x2_parts[1]) + conv_b, 0.2)
__global__ __launch_bounds__(256)
void k_conv(const float* __restrict__ mid1, const float* __restrict__ mid2,
            const float* __restrict__ convW, const float* __restrict__ convB,
            float* __restrict__ out) {
  int o   = blockIdx.x;        // 0..511
  int pos = threadIdx.x;       // b*16 + h*4 + w
  int b   = pos >> 4;
  int hw  = pos & 15;
  const float* wr = convW + (size_t)o * 1024;
  float s = 0.0f;
  for (int i = 0; i < 512; ++i) s = fmaf(wr[i],       mid1[(b * 512 + i) * 16 + hw], s);
  for (int i = 0; i < 512; ++i) s = fmaf(wr[512 + i], mid2[(b * 512 + i) * 16 + hw], s);
  s += convB[o];
  out[(b * 512 + o) * 16 + hw] = s > 0.0f ? s : 0.2f * s;
}

// ------------------------------- host --------------------------------------

extern "C" void kernel_launch(void* const* d_in, const int* in_sizes, int n_in,
                              void* d_out, int out_size, void* d_ws, size_t ws_size,
                              hipStream_t stream) {
  const float* x1     = (const float*)d_in[0];
  const float* x2     = (const float*)d_in[1];
  const float* encWih = (const float*)d_in[2];
  const float* encWhh = (const float*)d_in[3];
  const float* encBih = (const float*)d_in[4];
  const float* encBhh = (const float*)d_in[5];
  const float* decWih = (const float*)d_in[6];
  const float* decWhh = (const float*)d_in[7];
  const float* decBih = (const float*)d_in[8];
  const float* decBhh = (const float*)d_in[9];
  const float* convW  = (const float*)d_in[10];
  const float* convB  = (const float*)d_in[11];
  float* out = (float*)d_out;

  const size_t WN = (size_t)2 * kG * kL;  // elems per LSTM weight tensor

  // ---- workspace carve (all sizes multiples of 256B) ----
  char* wsb = (char*)d_ws;
  size_t off = 0;
  auto carve = [&](size_t bytes) -> void* {
    void* p = wsb + off;
    off += (bytes + 255) & ~(size_t)255;
    return p;
  };
  __bf16* wEih = (__bf16*)carve(WN * 2);
  __bf16* wEhh = (__bf16*)carve(WN * 2);
  __bf16* wDih = (__bf16*)carve(WN * 2);
  __bf16* wDhh = (__bf16*)carve(WN * 2);
  __bf16* hbuf[2][2];
  hbuf[0][0] = (__bf16*)carve((size_t)kBL * 2);
  hbuf[0][1] = (__bf16*)carve((size_t)kBL * 2);
  hbuf[1][0] = (__bf16*)carve((size_t)kBL * 2);
  hbuf[1][1] = (__bf16*)carve((size_t)kBL * 2);
  float* cbuf0 = (float*)carve((size_t)kBL * 4);
  float* cbuf1 = (float*)carve((size_t)kBL * 4);
  __bf16* xchunk = (__bf16*)carve((size_t)kBL * 2);
  float* saves = (float*)carve((size_t)16 * kBL * 4);  // 16 x [16,2048] fp32

  // ---- 1. weights fp32 -> bf16 (one-time 0.8 GB pass; halves step traffic)
  {
    int blocks = (int)((WN / 4 + 255) / 256);
    k_cvt_bf16<<<blocks, 256, 0, stream>>>(encWih, wEih, (int)WN);
    k_cvt_bf16<<<blocks, 256, 0, stream>>>(encWhh, wEhh, (int)WN);
    k_cvt_bf16<<<blocks, 256, 0, stream>>>(decWih, wDih, (int)WN);
    k_cvt_bf16<<<blocks, 256, 0, stream>>>(decWhh, wDhh, (int)WN);
  }

  // ---- LSTM sequencing ----
  int ping0 = 0, ping1 = 0;
  auto zero_state = [&]() {
    // hbuf[0][0]..cbuf1 are contiguous: 4*64KB + 2*128KB = 512KB = 131072 dw
    k_zero<<<(131072 + 255) / 256, 256, 0, stream>>>((uint32_t*)hbuf[0][0], 131072);
    ping0 = ping1 = 0;
  };

  const __bf16* last1 = hbuf[1][0];
  auto lstm2 = [&](const __bf16* xin, bool dec, float* save) {
    const __bf16* Wih = dec ? wDih : wEih;
    const __bf16* Whh = dec ? wDhh : wEhh;
    const float* bih  = dec ? decBih : encBih;
    const float* bhh  = dec ? decBhh : encBhh;
    // layer 0
    __bf16* h0o = hbuf[0][ping0 ^ 1];
    k_lstm_step<<<128, 256, 0, stream>>>(xin, hbuf[0][ping0], Wih, Whh,
                                         bih, bhh, cbuf0, h0o, nullptr);
    ping0 ^= 1;
    // layer 1 (input = layer-0 hn; weight/bias layer offset)
    __bf16* h1o = hbuf[1][ping1 ^ 1];
    k_lstm_step<<<128, 256, 0, stream>>>(h0o, hbuf[1][ping1],
                                         Wih + (size_t)kG * kL, Whh + (size_t)kG * kL,
                                         bih + kG, bhh + kG, cbuf1, h1o, save);
    ping1 ^= 1;
    last1 = h1o;
  };
  auto extract = [&](const float* x, int s) {
    k_extract<<<(kBL + 255) / 256, 256, 0, stream>>>(x, s, xchunk);
  };

  // Phase A: enc(x2) -> dec scan(x1) -> AR (de1 + 7 preds, save slots 0..7)
  zero_state();
  for (int s = 0; s < 4; ++s) { extract(x2, s); lstm2(xchunk, false, nullptr); }
  for (int s = 0; s < 4; ++s) {
    extract(x1, s);
    lstm2(xchunk, true, s == 3 ? saves : nullptr);                 // slot0 = de1
  }
  for (int t = 0; t < 7; ++t) lstm2(last1, true, saves + (size_t)(1 + t) * kBL);

  // Phase B: enc(x1 rev) -> dec scan(x2 rev) -> AR (de2 + 7 preds, slots 8..15)
  zero_state();
  for (int s = 3; s >= 0; --s) { extract(x1, s); lstm2(xchunk, false, nullptr); }
  for (int s = 3; s >= 0; --s) {
    extract(x2, s);
    lstm2(xchunk, true, s == 0 ? saves + (size_t)8 * kBL : nullptr);  // slot8 = de2
  }
  for (int t = 0; t < 7; ++t) lstm2(last1, true, saves + (size_t)(9 + t) * kBL);

  // ---- outputs: (conv, x1, x1p1, x1p2, x2p0, x2p1, x2) each kImg floats ----
  hipMemcpyAsync(out + (size_t)1 * kImg, x1, (size_t)kImg * 4,
                 hipMemcpyDeviceToDevice, stream);
  hipMemcpyAsync(out + (size_t)6 * kImg, x2, (size_t)kImg * 4,
                 hipMemcpyDeviceToDevice, stream);
  k_assemble<<<(4 * kImg + 255) / 256, 256, 0, stream>>>(saves, out);
  k_conv<<<512, 256, 0, stream>>>(out + (size_t)2 * kImg, out + (size_t)5 * kImg,
                                  convW, convB, out);
}
